// SlotAttention_21698174780001
// MI455X (gfx1250) — matrix operand order repaired and checked by reading, not compiled, not run on hardware
//
#include <hip/hip_runtime.h>
#include <hip/hip_bf16.h>

// ---------------- problem constants ----------------
#define NTOK   262144
#define CIN    512
#define DDIM   64
#define NSLOT  3
#define EPS_W  1e-8f
#define LN_EPS 1e-3f

typedef __bf16 bf16_t;
typedef __attribute__((ext_vector_type(8)))  __bf16 v8bf;
typedef __attribute__((ext_vector_type(16))) __bf16 v16bf;
typedef __attribute__((ext_vector_type(8)))  float  v8f;

// ================================================================
// Kernel 1: fused LayerNorm(x) + k = x_hat@Wk, v = x_hat@Wv  (WMMA bf16)
//   grid 2048 x 256 threads (8 waves), 128 rows / block, 16 rows / wave.
//   LDS: Wk^T, Wv^T in bf16 (pitch 520), ln_g/ln_b, per-wave bf16 A tile.
// ================================================================
#define WPITCH 520            // 512 + 8 pad (bf16 elements)
#define LDS_WTK 0
#define LDS_WTV 66560         // 64*520*2
#define LDS_G   133120
#define LDS_B   135168
#define LDS_A   137216        // + wave*16*520*2  (8 waves -> 133120 B)
#define LDS_TOTAL 270336

__global__ void __launch_bounds__(256) ln_kv_kernel(
    const float* __restrict__ x, const float* __restrict__ ln_g,
    const float* __restrict__ ln_b, const float* __restrict__ Wk,
    const float* __restrict__ Wv, float* __restrict__ kout,
    float* __restrict__ vout)
{
  extern __shared__ char smem[];
  bf16_t* WTk = (bf16_t*)(smem + LDS_WTK);
  bf16_t* WTv = (bf16_t*)(smem + LDS_WTV);
  float*  gls = (float*)(smem + LDS_G);
  float*  bls = (float*)(smem + LDS_B);

  const int t    = threadIdx.x;
  const int wave = t >> 5;
  const int lane = t & 31;
  const int r    = lane & 15;     // A row / B column / C column index
  const int kh   = lane >> 4;     // lane-half selects K sub-range

  bf16_t* Aw = (bf16_t*)(smem + LDS_A) + wave * (16 * WPITCH);

  // ---- cooperative weight transpose + convert to bf16 ----
  for (int i = t; i < CIN * DDIM; i += 256) {
    int K = i >> 6, n = i & 63;
    WTk[n * WPITCH + K] = (bf16_t)Wk[i];
    WTv[n * WPITCH + K] = (bf16_t)Wv[i];
  }
  for (int i = t; i < CIN; i += 256) { gls[i] = ln_g[i]; bls[i] = ln_b[i]; }
  __syncthreads();

  const size_t row = (size_t)blockIdx.x * 128 + (size_t)wave * 16 + r;
  const float* xr  = x + row * CIN;

  // ---- pass 1: LN statistics (lane covers 8-wide chunks of its half) ----
  float sum = 0.f, sq = 0.f;
  #pragma unroll
  for (int u = 0; u < 32; ++u) {
    int c = 16 * u + 8 * kh;
    float4 a0 = *(const float4*)(xr + c);
    float4 a1 = *(const float4*)(xr + c + 4);
    sum += a0.x + a0.y + a0.z + a0.w + a1.x + a1.y + a1.z + a1.w;
    sq  += a0.x*a0.x + a0.y*a0.y + a0.z*a0.z + a0.w*a0.w
         + a1.x*a1.x + a1.y*a1.y + a1.z*a1.z + a1.w*a1.w;
  }
  sum += __shfl_xor(sum, 16, 32);
  sq  += __shfl_xor(sq, 16, 32);
  const float mean = sum * (1.f / 512.f);
  const float rstd = rsqrtf(sq * (1.f / 512.f) - mean * mean + LN_EPS);

  // ---- pass 2: normalize, cvt bf16, stage A tile in LDS (L2 re-hit) ----
  #pragma unroll
  for (int u = 0; u < 32; ++u) {
    int c = 16 * u + 8 * kh;
    float4 a0 = *(const float4*)(xr + c);
    float4 a1 = *(const float4*)(xr + c + 4);
    float4 g0 = *(const float4*)(gls + c);
    float4 g1 = *(const float4*)(gls + c + 4);
    float4 b0 = *(const float4*)(bls + c);
    float4 b1 = *(const float4*)(bls + c + 4);
    v8bf o;
    o[0] = (bf16_t)((a0.x - mean) * rstd * g0.x + b0.x);
    o[1] = (bf16_t)((a0.y - mean) * rstd * g0.y + b0.y);
    o[2] = (bf16_t)((a0.z - mean) * rstd * g0.z + b0.z);
    o[3] = (bf16_t)((a0.w - mean) * rstd * g0.w + b0.w);
    o[4] = (bf16_t)((a1.x - mean) * rstd * g1.x + b1.x);
    o[5] = (bf16_t)((a1.y - mean) * rstd * g1.y + b1.y);
    o[6] = (bf16_t)((a1.z - mean) * rstd * g1.z + b1.z);
    o[7] = (bf16_t)((a1.w - mean) * rstd * g1.w + b1.w);
    *(v8bf*)(Aw + r * WPITCH + c) = o;
  }
  __syncthreads();

  // ---- WMMA GEMM: 16x64 output per wave per matrix, K = 512 ----
  const size_t rowBase = (size_t)blockIdx.x * 128 + (size_t)wave * 16;
  for (int mat = 0; mat < 2; ++mat) {
    const bf16_t* WT  = mat ? WTv : WTk;
    float* outp       = mat ? vout : kout;
    for (int nt = 0; nt < 4; ++nt) {
      v8f acc = {};
      const bf16_t* Bc = WT + (nt * 16 + r) * WPITCH + kh * 8;   // same K layout as A
      const bf16_t* Ar = Aw + r * WPITCH + kh * 8;               // A lane layout
      #pragma unroll
      for (int kk = 0; kk < 16; ++kk) {
        v8bf alo = *(const v8bf*)(Ar + kk * 32);        // K = 32kk+8kh .. +7
        v8bf ahi = *(const v8bf*)(Ar + kk * 32 + 16);   // K = 32kk+16+8kh .. +7
        v8bf blo = *(const v8bf*)(Bc + kk * 32);        // K = 32kk+8kh .. +7
        v8bf bhi = *(const v8bf*)(Bc + kk * 32 + 16);   // K = 32kk+16+8kh .. +7
        v16bf av = __builtin_shufflevector(alo, ahi, 0,1,2,3,4,5,6,7,8,9,10,11,12,13,14,15);
        v16bf bv = __builtin_shufflevector(blo, bhi, 0,1,2,3,4,5,6,7,8,9,10,11,12,13,14,15);
        acc = __builtin_amdgcn_wmma_f32_16x16x32_bf16(
            false, av, false, bv, (short)0, acc, false, false);
      }
      #pragma unroll
      for (int i2 = 0; i2 < 8; ++i2) {                  // C: M=i2+8*kh, N=r
        size_t m = rowBase + (size_t)(i2 + 8 * kh);
        outp[m * DDIM + nt * 16 + r] = acc[i2];
      }
    }
  }
}

// ================================================================
// Kernel 2: attention pass. grid 1024 x 256. One row per thread for the
// k.q dots + softmax; wave-cooperative (p+eps)*v reduction; per-block
// partials (192 U + 3 S) written for a deterministic 2nd-stage reduce.
// ================================================================
__global__ void __launch_bounds__(256) attend_kernel(
    const float* __restrict__ k, const float* __restrict__ v,
    const float* __restrict__ qbuf, float* __restrict__ partial,
    float* __restrict__ attn_out, int write_attn)
{
  __shared__ float qs[192];
  __shared__ float uacc[8][192];
  __shared__ float sacc[8][3];
  const int t = threadIdx.x;
  const int lane = t & 31, wave = t >> 5;
  if (t < 192) qs[t] = qbuf[t];
  __syncthreads();

  const size_t i = (size_t)blockIdx.x * 256 + t;
  const float4* kr = (const float4*)(k + i * DDIM);
  const float4* q4 = (const float4*)qs;
  float l0 = 0.f, l1 = 0.f, l2 = 0.f;
  #pragma unroll
  for (int j = 0; j < 16; ++j) {
    float4 kv = kr[j];
    float4 q0 = q4[j], q1 = q4[16 + j], q2 = q4[32 + j];
    l0 += kv.x*q0.x + kv.y*q0.y + kv.z*q0.z + kv.w*q0.w;
    l1 += kv.x*q1.x + kv.y*q1.y + kv.z*q1.z + kv.w*q1.w;
    l2 += kv.x*q2.x + kv.y*q2.y + kv.z*q2.z + kv.w*q2.w;
  }
  float m  = fmaxf(l0, fmaxf(l1, l2));
  float e0 = expf(l0 - m), e1 = expf(l1 - m), e2 = expf(l2 - m);
  float inv = 1.f / (e0 + e1 + e2);
  float p0 = e0 * inv, p1 = e1 * inv, p2 = e2 * inv;
  if (write_attn) {                    // attn.T rows: [bg, fg0, fg1]
    attn_out[i] = p0;
    attn_out[(size_t)NTOK + i] = p1;
    attn_out[2 * (size_t)NTOK + i] = p2;
  }
  float pe0 = p0 + EPS_W, pe1 = p1 + EPS_W, pe2 = p2 + EPS_W;

  float s0 = pe0, s1 = pe1, s2 = pe2;
  #pragma unroll
  for (int off = 16; off > 0; off >>= 1) {
    s0 += __shfl_xor(s0, off, 32);
    s1 += __shfl_xor(s1, off, 32);
    s2 += __shfl_xor(s2, off, 32);
  }

  // wave-cooperative U += (p+eps) * v : lane owns columns 2*lane, 2*lane+1
  float2 a0 = {0.f, 0.f}, a1 = {0.f, 0.f}, a2 = {0.f, 0.f};
  const float* vb = v + ((size_t)blockIdx.x * 256 + (size_t)wave * 32) * DDIM + 2 * lane;
  for (int rr = 0; rr < 32; ++rr) {
    float w0 = __shfl(pe0, rr, 32);
    float w1 = __shfl(pe1, rr, 32);
    float w2 = __shfl(pe2, rr, 32);
    float2 vv = *(const float2*)(vb + (size_t)rr * DDIM);
    a0.x += w0 * vv.x; a0.y += w0 * vv.y;
    a1.x += w1 * vv.x; a1.y += w1 * vv.y;
    a2.x += w2 * vv.x; a2.y += w2 * vv.y;
  }
  uacc[wave][      2*lane] = a0.x;  uacc[wave][      2*lane+1] = a0.y;
  uacc[wave][ 64 + 2*lane] = a1.x;  uacc[wave][ 64 + 2*lane+1] = a1.y;
  uacc[wave][128 + 2*lane] = a2.x;  uacc[wave][128 + 2*lane+1] = a2.y;
  if (lane == 0) { sacc[wave][0] = s0; sacc[wave][1] = s1; sacc[wave][2] = s2; }
  __syncthreads();

  if (t < 192) {
    float s = 0.f;
    #pragma unroll
    for (int w = 0; w < 8; ++w) s += uacc[w][t];
    partial[(size_t)blockIdx.x * 256 + t] = s;
  } else if (t < 195) {
    int s3 = t - 192;
    float s = 0.f;
    #pragma unroll
    for (int w = 0; w < 8; ++w) s += sacc[w][s3];
    partial[(size_t)blockIdx.x * 256 + t] = s;
  }
}

// ================================================================
// Kernel 3: deterministic fixed-order reduction of 1024 block partials.
// ================================================================
__global__ void __launch_bounds__(256) reduce_kernel(
    const float* __restrict__ partial, float* __restrict__ Ured)
{
  int t = threadIdx.x;
  if (t >= 195) return;
  float s = 0.f;
  for (int b = 0; b < 1024; ++b) s += partial[(size_t)b * 256 + t];
  Ured[t] = s;
}

// ---------------- small LN+matmul helper on 64-thread block ----------------
__device__ __forceinline__ float ln64(float val, const float* buf,
                                      const float* g, const float* b, int t)
{
  float mean = 0.f;
  for (int d = 0; d < 64; ++d) mean += buf[d];
  mean *= (1.f / 64.f);
  float var = 0.f;
  for (int d = 0; d < 64; ++d) { float dd = buf[d] - mean; var += dd * dd; }
  var *= (1.f / 64.f);
  return (val - mean) * rsqrtf(var + LN_EPS) * g[t] + b[t];
}

// ================================================================
// Kernel 4: slot init (mu + exp(ls)*noise) and q0 computation. <<<1,64>>>
// ================================================================
__global__ void __launch_bounds__(64) init_kernel(
    const float* __restrict__ noise_fg, const float* __restrict__ noise_bg,
    const float* __restrict__ mu_fg, const float* __restrict__ ls_fg,
    const float* __restrict__ mu_bg, const float* __restrict__ ls_bg,
    const float* __restrict__ qfg_g, const float* __restrict__ qfg_b,
    const float* __restrict__ qfg_W, const float* __restrict__ qbg_g,
    const float* __restrict__ qbg_b, const float* __restrict__ qbg_W,
    float* __restrict__ slots, float* __restrict__ qbuf)
{
  __shared__ float tmp[64], tmp2[64];
  const int t = threadIdx.x;
  for (int s = 0; s < NSLOT; ++s) {
    float sv = (s == 0) ? (mu_bg[t] + expf(ls_bg[t]) * noise_bg[t])
                        : (mu_fg[t] + expf(ls_fg[t]) * noise_fg[(s - 1) * 64 + t]);
    slots[s * 64 + t] = sv;
    const float* qg = (s == 0) ? qbg_g : qfg_g;
    const float* qb = (s == 0) ? qbg_b : qfg_b;
    const float* qW = (s == 0) ? qbg_W : qfg_W;
    tmp[t] = sv; __syncthreads();
    float lq = ln64(sv, tmp, qg, qb, t);
    tmp2[t] = lq; __syncthreads();
    float qv = 0.f;
    for (int d = 0; d < 64; ++d) qv += tmp2[d] * qW[d * 64 + t];
    qbuf[s * 64 + t] = qv * 0.125f;          // scale = D^-0.5
    __syncthreads();
  }
}

// ================================================================
// Kernel 5: slot update: upd = U/S, GRU, +MLP(LN), next-q. <<<1,64>>>
// ================================================================
__global__ void __launch_bounds__(64) update_kernel(
    const float* __restrict__ Ured, float* __restrict__ slots,
    float* __restrict__ qbuf,
    const float* __restrict__ grufg_k, const float* __restrict__ grufg_rk,
    const float* __restrict__ grufg_bias,
    const float* __restrict__ grubg_k, const float* __restrict__ grubg_rk,
    const float* __restrict__ grubg_bias,
    const float* __restrict__ mlpfg_g, const float* __restrict__ mlpfg_b,
    const float* __restrict__ mlpfg_W1, const float* __restrict__ mlpfg_b1,
    const float* __restrict__ mlpfg_W2, const float* __restrict__ mlpfg_b2,
    const float* __restrict__ mlpbg_g, const float* __restrict__ mlpbg_b,
    const float* __restrict__ mlpbg_W1, const float* __restrict__ mlpbg_b1,
    const float* __restrict__ mlpbg_W2, const float* __restrict__ mlpbg_b2,
    const float* __restrict__ qfg_g, const float* __restrict__ qfg_b,
    const float* __restrict__ qfg_W, const float* __restrict__ qbg_g,
    const float* __restrict__ qbg_b, const float* __restrict__ qbg_W,
    float* __restrict__ slots_out, int last)
{
  __shared__ float u[64], h[64], tmp[64], tmp2[64];
  const int t = threadIdx.x;
  for (int s = 0; s < NSLOT; ++s) {
    const int bg = (s == 0);
    const float* gk  = bg ? grubg_k    : grufg_k;
    const float* grk = bg ? grubg_rk   : grufg_rk;
    const float* gbi = bg ? grubg_bias : grufg_bias;
    const float* mg  = bg ? mlpbg_g  : mlpfg_g;
    const float* mb  = bg ? mlpbg_b  : mlpfg_b;
    const float* W1  = bg ? mlpbg_W1 : mlpfg_W1;
    const float* b1  = bg ? mlpbg_b1 : mlpfg_b1;
    const float* W2  = bg ? mlpbg_W2 : mlpfg_W2;
    const float* b2  = bg ? mlpbg_b2 : mlpfg_b2;
    const float* qg  = bg ? qbg_g : qfg_g;
    const float* qb  = bg ? qbg_b : qfg_b;
    const float* qW  = bg ? qbg_W : qfg_W;

    const float S = Ured[192 + s];
    u[t] = Ured[s * 64 + t] / S;
    h[t] = slots[s * 64 + t];
    __syncthreads();

    float xm[3], hm[3];
    #pragma unroll
    for (int c = 0; c < 3; ++c) {
      int j = c * 64 + t;
      float ax = gbi[j];          // bias[0]
      float ah = gbi[192 + j];    // bias[1]
      for (int d = 0; d < 64; ++d) {
        ax += u[d] * gk[d * 192 + j];
        ah += h[d] * grk[d * 192 + j];
      }
      xm[c] = ax; hm[c] = ah;
    }
    float z    = 1.f / (1.f + expf(-(xm[0] + hm[0])));
    float rg   = 1.f / (1.f + expf(-(xm[1] + hm[1])));
    float cand = tanhf(xm[2] + rg * hm[2]);
    float hn   = z * h[t] + (1.f - z) * cand;
    __syncthreads();
    h[t] = hn; __syncthreads();

    // residual MLP with pre-LN
    float lnv = ln64(hn, h, mg, mb, t);
    tmp[t] = lnv; __syncthreads();
    float h1 = b1[t];
    for (int d = 0; d < 64; ++d) h1 += tmp[d] * W1[d * 64 + t];
    h1 = fmaxf(h1, 0.f);
    tmp2[t] = h1; __syncthreads();
    float o = b2[t];
    for (int d = 0; d < 64; ++d) o += tmp2[d] * W2[d * 64 + t];
    float snew = hn + o;
    slots[s * 64 + t] = snew;
    if (last) slots_out[s * 64 + t] = snew;

    // q for next iteration
    __syncthreads(); tmp[t] = snew; __syncthreads();
    float lq = ln64(snew, tmp, qg, qb, t);
    tmp2[t] = lq; __syncthreads();
    float qv = 0.f;
    for (int d = 0; d < 64; ++d) qv += tmp2[d] * qW[d * 64 + t];
    qbuf[s * 64 + t] = qv * 0.125f;
    __syncthreads();
  }
}

// ================================================================
extern "C" void kernel_launch(void* const* d_in, const int* in_sizes, int n_in,
                              void* d_out, int out_size, void* d_ws, size_t ws_size,
                              hipStream_t stream) {
  (void)in_sizes; (void)n_in; (void)out_size; (void)ws_size;
  const float* x        = (const float*)d_in[0];
  const float* noise_fg = (const float*)d_in[1];
  const float* noise_bg = (const float*)d_in[2];
  const float* ln_g     = (const float*)d_in[3];
  const float* ln_b     = (const float*)d_in[4];
  const float* mu_fg    = (const float*)d_in[5];
  const float* ls_fg    = (const float*)d_in[6];
  const float* mu_bg    = (const float*)d_in[7];
  const float* ls_bg    = (const float*)d_in[8];
  const float* Wk       = (const float*)d_in[9];
  const float* Wv       = (const float*)d_in[10];
  const float* qfg_g    = (const float*)d_in[11];
  const float* qfg_b    = (const float*)d_in[12];
  const float* qfg_W    = (const float*)d_in[13];
  const float* qbg_g    = (const float*)d_in[14];
  const float* qbg_b    = (const float*)d_in[15];
  const float* qbg_W    = (const float*)d_in[16];
  const float* grufg_k    = (const float*)d_in[17];
  const float* grufg_rk   = (const float*)d_in[18];
  const float* grufg_bias = (const float*)d_in[19];
  const float* grubg_k    = (const float*)d_in[20];
  const float* grubg_rk   = (const float*)d_in[21];
  const float* grubg_bias = (const float*)d_in[22];
  const float* mlpfg_g  = (const float*)d_in[23];
  const float* mlpfg_b  = (const float*)d_in[24];
  const float* mlpfg_W1 = (const float*)d_in[25];
  const float* mlpfg_b1 = (const float*)d_in[26];
  const float* mlpfg_W2 = (const float*)d_in[27];
  const float* mlpfg_b2 = (const float*)d_in[28];
  const float* mlpbg_g  = (const float*)d_in[29];
  const float* mlpbg_b  = (const float*)d_in[30];
  const float* mlpbg_W1 = (const float*)d_in[31];
  const float* mlpbg_b1 = (const float*)d_in[32];
  const float* mlpbg_W2 = (const float*)d_in[33];
  const float* mlpbg_b2 = (const float*)d_in[34];

  float* ws      = (float*)d_ws;
  float* kbuf    = ws;                                   // N*64
  float* vbuf    = kbuf + (size_t)NTOK * DDIM;           // N*64
  float* partial = vbuf + (size_t)NTOK * DDIM;           // 1024*256
  float* Ured    = partial + (size_t)1024 * 256;         // 256
  float* slots   = Ured + 256;                           // 192
  float* qbuf    = slots + 192;                          // 192

  float* out      = (float*)d_out;                       // [0,192): slots
  float* attn_out = out + 192;                           // [192, 192+3N): attn.T

  ln_kv_kernel<<<NTOK / 128, 256, LDS_TOTAL, stream>>>(
      x, ln_g, ln_b, Wk, Wv, kbuf, vbuf);
  init_kernel<<<1, 64, 0, stream>>>(
      noise_fg, noise_bg, mu_fg, ls_fg, mu_bg, ls_bg,
      qfg_g, qfg_b, qfg_W, qbg_g, qbg_b, qbg_W, slots, qbuf);

  for (int it = 0; it < 2; ++it) {
    int last = (it == 1);
    attend_kernel<<<NTOK / 256, 256, 0, stream>>>(
        kbuf, vbuf, qbuf, partial, attn_out, last);
    reduce_kernel<<<1, 256, 0, stream>>>(partial, Ured);
    update_kernel<<<1, 64, 0, stream>>>(
        Ured, slots, qbuf,
        grufg_k, grufg_rk, grufg_bias, grubg_k, grubg_rk, grubg_bias,
        mlpfg_g, mlpfg_b, mlpfg_W1, mlpfg_b1, mlpfg_W2, mlpfg_b2,
        mlpbg_g, mlpbg_b, mlpbg_W1, mlpbg_b1, mlpbg_W2, mlpbg_b2,
        qfg_g, qfg_b, qfg_W, qbg_g, qbg_b, qbg_W,
        out, last);
  }
}
